// SpikeSA_75720273429141
// MI455X (gfx1250) — matrix operand order repaired and checked by reading, not compiled, not run on hardware
//
#include <hip/hip_runtime.h>

// ---------------------------------------------------------------------------
// SpikeSA for MI455X (gfx1250).
//   Pass 1: split fp32 -> bf16 hi/lo planes (x and the 4 weights), one pass.
//   Pass 2: 3x GEMM [65536,512]x[512,512] with split-bf16 WMMA (hi*hi+hi*lo+
//           lo*hi ~ fp32), tiles staged to LDS by the Tensor Data Mover
//           (tensor_load_to_lds, double buffered, padded LDS rows).
//   Pass 3: spiking attention (VALU), emits hi/lo bf16 planes in-place over
//           the x planes.
//   Pass 4: O-projection GEMM from those planes -> fp32 output.
// Wave tile 64x32 -> 24 WMMA : 24 ds_load_b128 per K-step per wave.
// ---------------------------------------------------------------------------

typedef __attribute__((ext_vector_type(16))) __bf16        v16bf;
typedef __attribute__((ext_vector_type(8)))  float         v8f;
typedef __attribute__((ext_vector_type(4)))  unsigned int  v4u;
typedef __attribute__((ext_vector_type(8)))  int           v8i;
typedef __attribute__((ext_vector_type(4)))  int           v4i;

struct U8x32 { uint4 a, b; };  // 32 bytes == sizeof(v16bf)

__device__ __forceinline__ unsigned short f2bf_rne(float f) {
  unsigned int u = __float_as_uint(f);
  u += 0x7FFFu + ((u >> 16) & 1u);
  return (unsigned short)(u >> 16);
}
__device__ __forceinline__ float bf2f(unsigned short h) {
  return __uint_as_float(((unsigned int)h) << 16);
}

// ---------------------------------------------------------------------------
// Pass 1: fp32[n] -> bf16 hi[n], lo[n]  (8 elements per thread)
// ---------------------------------------------------------------------------
__global__ __launch_bounds__(256) void split_fp32_bf16(
    const float* __restrict__ src, unsigned short* __restrict__ hi,
    unsigned short* __restrict__ lo, long n)
{
  long i = ((long)blockIdx.x * 256 + threadIdx.x) * 8;
  if (i >= n) return;
  float4 a = *(const float4*)&src[i];
  float4 b = *(const float4*)&src[i + 4];
  float v[8] = {a.x, a.y, a.z, a.w, b.x, b.y, b.z, b.w};
  union { unsigned short s[8]; uint4 q; } H, L;
#pragma unroll
  for (int j = 0; j < 8; ++j) {
    unsigned short h = f2bf_rne(v[j]);
    H.s[j] = h;
    L.s[j] = f2bf_rne(v[j] - bf2f(h));
  }
  *(uint4*)&hi[i] = H.q;
  *(uint4*)&lo[i] = L.q;
}

// ---------------------------------------------------------------------------
// WMMA fragment gathers from padded LDS rows (stride LDSK ushorts = 80B)
// ---------------------------------------------------------------------------
__device__ __forceinline__ v16bf load_fragA(const unsigned short* rowp, int lh) {
  U8x32 u;
  const char* p = (const char*)rowp + lh * 16;
  u.a = *(const uint4*)(p);
  u.b = *(const uint4*)(p + 32);
  return __builtin_bit_cast(v16bf, u);
}
__device__ __forceinline__ v16bf load_fragB(const unsigned short* rowp, int lh) {
  U8x32 u;
  const char* p = (const char*)rowp + lh * 16;
  u.a = *(const uint4*)(p);
  u.b = *(const uint4*)(p + 32);
  return __builtin_bit_cast(v16bf, u);
}

#define BM 128
#define BN 128
#define BK 32
#define LDSK 40   // 80B padded LDS row stride (64B data + 16B pad from TDM)

// TDM: 2D tile (BK x tile_rows) of bf16, row-major [rows][K], into LDS with
// padding: 16 DWORDs (one 32-elem bf16 row) then +4 DWORDs -> 80B row stride.
// D# layout per CDNA5 ISA 8.3/8.4. 2D: groups 2/3 (and extra group) zero.
__device__ __forceinline__ void tdm_load_2d(unsigned lds_off, const void* gptr,
                                            unsigned tile_rows, unsigned tensor_rows,
                                            unsigned K)
{
  unsigned long long ga = (unsigned long long)(size_t)gptr;
  v4u g0;
  g0[0] = 1u;                                      // count=1, user ctrl bits 0
  g0[1] = lds_off;                                 // lds_addr (bytes)
  g0[2] = (unsigned)(ga & 0xFFFFFFFFull);          // global_addr[31:0]
  g0[3] = (unsigned)((ga >> 32) & 0x01FFFFFFull) | (2u << 30);  // type=2
  v8i g1;
  // data_size=2B(1), pad_enable, pad_interval=16dw(3), pad_amount=4dw(3)
  g1[0] = (int)((1u << 16) | (1u << 20) | (3u << 22) | (3u << 25));
  g1[1] = (int)((K & 0xFFFFu) << 16);                       // tensor_dim0 lo16
  g1[2] = (int)((K >> 16) | ((tensor_rows & 0xFFFFu) << 16));
  g1[3] = (int)((tensor_rows >> 16) | ((unsigned)BK << 16)); // tile_dim0=BK
  g1[4] = (int)tile_rows;                                    // tile_dim1
  g1[5] = (int)K;                                            // dim0 stride lo32
  g1[6] = 0;
  g1[7] = 0;
  v4i z4 = {0, 0, 0, 0};
  v8i z8 = {0, 0, 0, 0, 0, 0, 0, 0};
  __builtin_amdgcn_tensor_load_to_lds(g0, g1, z4, z4, z8, 0);
}

// ---------------------------------------------------------------------------
// Y[M,N] = (Xh+Xl)[M,K] * ((Wh+Wl)[N,K])^T + bias, dropping lo*lo.
// 8 waves: 2 (M) x 4 (N); wave tile 64x32 = 4x2 16x16 accum tiles.
// ---------------------------------------------------------------------------
__global__ __launch_bounds__(256) void gemm_bf16x3_tdm(
    const unsigned short* __restrict__ Xh, const unsigned short* __restrict__ Xl,
    const unsigned short* __restrict__ Wh, const unsigned short* __restrict__ Wl,
    const float* __restrict__ bias, float* __restrict__ Y,
    int M, int N, int K)
{
  __shared__ __align__(16) unsigned short sXh[2][BM][LDSK];
  __shared__ __align__(16) unsigned short sXl[2][BM][LDSK];
  __shared__ __align__(16) unsigned short sWh[2][BN][LDSK];
  __shared__ __align__(16) unsigned short sWl[2][BN][LDSK];

  const int tid  = threadIdx.x;
  const int lane = tid & 31;
  const int wave = tid >> 5;      // 0..7
  const int wm   = wave & 1;      // 0..1  (M direction, 64 rows each)
  const int wn   = wave >> 1;     // 0..3  (N direction, 32 cols each)
  const int lh   = lane >> 4;
  const int lr   = lane & 15;

  const int n0 = blockIdx.x * BN;
  const int m0 = blockIdx.y * BM;

  const unsigned short* gXh = Xh + (size_t)m0 * K;
  const unsigned short* gXl = Xl + (size_t)m0 * K;
  const unsigned short* gWh = Wh + (size_t)n0 * K;
  const unsigned short* gWl = Wl + (size_t)n0 * K;

  const v8f zacc = {0.f, 0.f, 0.f, 0.f, 0.f, 0.f, 0.f, 0.f};
  v8f acc[4][2];
#pragma unroll
  for (int i = 0; i < 4; ++i)
#pragma unroll
    for (int j = 0; j < 2; ++j) acc[i][j] = zacc;

  const int nk = K / BK;

  if (wave == 0) {   // uniform per wave: only wave 0 drives the TDM
    tdm_load_2d((unsigned)(size_t)&sXh[0][0][0], gXh, BM, (unsigned)M, (unsigned)K);
    tdm_load_2d((unsigned)(size_t)&sXl[0][0][0], gXl, BM, (unsigned)M, (unsigned)K);
    tdm_load_2d((unsigned)(size_t)&sWh[0][0][0], gWh, BN, (unsigned)N, (unsigned)K);
    tdm_load_2d((unsigned)(size_t)&sWl[0][0][0], gWl, BN, (unsigned)N, (unsigned)K);
  }

  for (int ik = 0; ik < nk; ++ik) {
    if (wave == 0) {
      if (ik + 1 < nk) {           // prefetch next tile into the other buffer
        const int nb = (ik + 1) & 1;
        const int ko = (ik + 1) * BK;
        tdm_load_2d((unsigned)(size_t)&sXh[nb][0][0], gXh + ko, BM, (unsigned)M, (unsigned)K);
        tdm_load_2d((unsigned)(size_t)&sXl[nb][0][0], gXl + ko, BM, (unsigned)M, (unsigned)K);
        tdm_load_2d((unsigned)(size_t)&sWh[nb][0][0], gWh + ko, BN, (unsigned)N, (unsigned)K);
        tdm_load_2d((unsigned)(size_t)&sWl[nb][0][0], gWl + ko, BN, (unsigned)N, (unsigned)K);
        __builtin_amdgcn_s_wait_tensorcnt(4);   // current buffer complete
      } else {
        __builtin_amdgcn_s_wait_tensorcnt(0);
      }
    }
    __syncthreads();

    const int b = ik & 1;
    v16bf Bh[2], Bl[2];
#pragma unroll
    for (int ni = 0; ni < 2; ++ni) {
      const int r = wn * 32 + ni * 16 + lr;
      Bh[ni] = load_fragB(&sWh[b][r][0], lh);
      Bl[ni] = load_fragB(&sWl[b][r][0], lh);
    }
#pragma unroll
    for (int mi = 0; mi < 4; ++mi) {
      const int r = wm * 64 + mi * 16 + lr;
      v16bf Ah = load_fragA(&sXh[b][r][0], lh);
      v16bf Al = load_fragA(&sXl[b][r][0], lh);
#pragma unroll
      for (int ni = 0; ni < 2; ++ni) {
        acc[mi][ni] = __builtin_amdgcn_wmma_f32_16x16x32_bf16(
            false, Ah, false, Bh[ni], (short)0, acc[mi][ni], false, false);
        acc[mi][ni] = __builtin_amdgcn_wmma_f32_16x16x32_bf16(
            false, Ah, false, Bl[ni], (short)0, acc[mi][ni], false, false);
        acc[mi][ni] = __builtin_amdgcn_wmma_f32_16x16x32_bf16(
            false, Al, false, Bh[ni], (short)0, acc[mi][ni], false, false);
      }
    }
    __syncthreads();   // everyone done reading buf b before TDM refills it
  }

  // D layout: row = r + 8*lanehalf, col = lane%16
#pragma unroll
  for (int mi = 0; mi < 4; ++mi)
#pragma unroll
    for (int ni = 0; ni < 2; ++ni) {
      const int n = n0 + wn * 32 + ni * 16 + lr;
      const float bv = bias[n];
      const int mbase = m0 + wm * 64 + mi * 16 + lh * 8;
#pragma unroll
      for (int r = 0; r < 8; ++r)
        Y[(size_t)(mbase + r) * N + n] = acc[mi][ni][r] + bv;
    }
}

// ---------------------------------------------------------------------------
// Spiking attention: one wave per (b,n,h); q,k,v fp32 (T,B,N,C).
// spike(attn) <=> raw dot >= 16. Output written as bf16 hi/lo planes.
// ---------------------------------------------------------------------------
__global__ __launch_bounds__(256) void spike_attn(
    const float* __restrict__ Q, const float* __restrict__ Kb,
    const float* __restrict__ V,
    unsigned short* __restrict__ Oh, unsigned short* __restrict__ Ol,
    int B, int N, int H, int C)
{
  __shared__ __align__(16) float sq[8][4][64];
  __shared__ __align__(16) float sk[8][4][64];
  __shared__ __align__(16) float sv[8][4][64];

  const int tid  = threadIdx.x;
  const int wave = tid >> 5;
  const int lane = tid & 31;

  const long g = (long)blockIdx.x * 8 + wave;
  const int h = (int)(g % H);
  const long bn = g / H;
  const int n = (int)(bn % N);
  const int b = (int)(bn / N);

  const size_t base = ((size_t)b * N + n) * C + (size_t)h * 64;
  const size_t tstr = (size_t)B * N * C;

  {
    const int t = lane >> 3;
    const int c = (lane & 7) * 8;
    const size_t off = base + (size_t)t * tstr + c;
    *(float4*)&sq[wave][t][c]     = *(const float4*)&Q[off];
    *(float4*)&sq[wave][t][c + 4] = *(const float4*)&Q[off + 4];
    *(float4*)&sk[wave][t][c]     = *(const float4*)&Kb[off];
    *(float4*)&sk[wave][t][c + 4] = *(const float4*)&Kb[off + 4];
    *(float4*)&sv[wave][t][c]     = *(const float4*)&V[off];
    *(float4*)&sv[wave][t][c + 4] = *(const float4*)&V[off + 4];
  }
  __syncthreads();

  const int p  = lane & 15;
  const int tq = p >> 2;
  const int tu = p & 3;
  const int d0 = (lane >> 4) * 32;
  float s = 0.f;
  const float* qr = sq[wave][tq];
  const float* kr = sk[wave][tu];
#pragma unroll
  for (int j = 0; j < 32; ++j) s += qr[d0 + j] * kr[d0 + j];
  s += __shfl_xor(s, 16, 32);

  const int d = lane * 2;
#pragma unroll
  for (int t = 0; t < 4; ++t) {
    float o0 = 0.f, o1 = 0.f;
#pragma unroll
    for (int u = 0; u < 4; ++u) {
      const float su = __shfl(s, t * 4 + u, 32);
      if (su >= 16.0f) { o0 += sv[wave][u][d]; o1 += sv[wave][u][d + 1]; }
    }
    const size_t off = base + (size_t)t * tstr + d;
    unsigned short h0 = f2bf_rne(o0), h1 = f2bf_rne(o1);
    unsigned short l0 = f2bf_rne(o0 - bf2f(h0));
    unsigned short l1 = f2bf_rne(o1 - bf2f(h1));
    *(unsigned*)&Oh[off] = (unsigned)h0 | ((unsigned)h1 << 16);
    *(unsigned*)&Ol[off] = (unsigned)l0 | ((unsigned)l1 << 16);
  }
}

extern "C" void kernel_launch(void* const* d_in, const int* in_sizes, int n_in,
                              void* d_out, int out_size, void* d_ws, size_t ws_size,
                              hipStream_t stream) {
  (void)in_sizes; (void)n_in; (void)out_size; (void)ws_size;
  const float* x  = (const float*)d_in[0];
  const float* Wq = (const float*)d_in[1];
  const float* bq = (const float*)d_in[2];
  const float* Wk = (const float*)d_in[3];
  const float* bk = (const float*)d_in[4];
  const float* Wv = (const float*)d_in[5];
  const float* bv = (const float*)d_in[6];
  const float* Wo = (const float*)d_in[7];
  const float* bo = (const float*)d_in[8];
  float* out = (float*)d_out;

  const int T = 4, B = 4, N = 4096, C = 512, H = 8;
  const int M = T * B * N;                  // 65536
  const size_t S  = (size_t)M * C;          // 33,554,432 activation elems
  const size_t WS = (size_t)C * C;          // 262,144 weight elems

  // workspace carve (bytes): q/k/v fp32 | x hi/lo planes | 8 weight planes
  char* ws = (char*)d_ws;
  float* qb = (float*)ws;
  float* kb = qb + S;
  float* vb = kb + S;
  unsigned short* Xh = (unsigned short*)(ws + 3 * S * sizeof(float));
  unsigned short* Xl = Xh + S;
  unsigned short* wp = Xl + S;
  unsigned short *Wqh = wp + 0*WS, *Wql = wp + 1*WS;
  unsigned short *Wkh = wp + 2*WS, *Wkl = wp + 3*WS;
  unsigned short *Wvh = wp + 4*WS, *Wvl = wp + 5*WS;
  unsigned short *Woh = wp + 6*WS, *Wol = wp + 7*WS;

  dim3 blk(256);

  // Pass 1: one-time hi/lo splits
  hipLaunchKernelGGL(split_fp32_bf16, dim3((unsigned)(S / 2048)), blk, 0, stream, x,  Xh,  Xl,  (long)S);
  hipLaunchKernelGGL(split_fp32_bf16, dim3((unsigned)(WS / 2048)), blk, 0, stream, Wq, Wqh, Wql, (long)WS);
  hipLaunchKernelGGL(split_fp32_bf16, dim3((unsigned)(WS / 2048)), blk, 0, stream, Wk, Wkh, Wkl, (long)WS);
  hipLaunchKernelGGL(split_fp32_bf16, dim3((unsigned)(WS / 2048)), blk, 0, stream, Wv, Wvh, Wvl, (long)WS);
  hipLaunchKernelGGL(split_fp32_bf16, dim3((unsigned)(WS / 2048)), blk, 0, stream, Wo, Woh, Wol, (long)WS);

  // Pass 2: q/k/v projections (TDM-staged WMMA GEMM)
  dim3 gg(C / BN, M / BM);                  // (4, 512)
  hipLaunchKernelGGL(gemm_bf16x3_tdm, gg, blk, 0, stream, Xh, Xl, Wqh, Wql, bq, qb, M, C, C);
  hipLaunchKernelGGL(gemm_bf16x3_tdm, gg, blk, 0, stream, Xh, Xl, Wkh, Wkl, bk, kb, M, C, C);
  hipLaunchKernelGGL(gemm_bf16x3_tdm, gg, blk, 0, stream, Xh, Xl, Wvh, Wvl, bv, vb, M, C, C);

  // Pass 3: spiking attention; x planes are dead now -> reuse as O planes
  dim3 ga((unsigned)((size_t)B * N * H / 8));
  hipLaunchKernelGGL(spike_attn, ga, blk, 0, stream, qb, kb, vb, Xh, Xl, B, N, H, C);

  // Pass 4: output projection from the attention planes
  hipLaunchKernelGGL(gemm_bf16x3_tdm, gg, blk, 0, stream, Xh, Xl, Woh, Wol, bo, out, M, C, C);
}
